// NeuralODE_1932735283453
// MI455X (gfx1250) — compile-verified, matrix operand order checked by reading, not run on hardware
//
#include <hip/hip_runtime.h>
#include <hip/hip_bf16.h>
#include <math.h>

// Problem constants (from reference)
#define BS   512
#define DD   1024
#define HIDD 2048
#define TLEN 11
#define NSTEPS ((TLEN - 1) * 2)

typedef __attribute__((ext_vector_type(16))) __bf16       v16bf;
typedef __attribute__((ext_vector_type(2)))  __bf16       v2bf;
typedef __attribute__((ext_vector_type(8)))  float        v8f;
typedef __attribute__((ext_vector_type(4)))  unsigned int u32x4;

// GCC-style vector to match the async builtin's parameter type exactly.
typedef int v4i_gcc __attribute__((__vector_size__(16)));
typedef __attribute__((address_space(1))) v4i_gcc* gptr_v4i;
typedef __attribute__((address_space(3))) v4i_gcc* lptr_v4i;

// ---------------- fast conversions ----------------------------------------
static __device__ __forceinline__ unsigned int pack2_bf16(float a, float b) {
#if __has_builtin(__builtin_amdgcn_cvt_pk_bf16_f32)
  v2bf r = __builtin_amdgcn_cvt_pk_bf16_f32(a, b);
#else
  v2bf r = { (__bf16)a, (__bf16)b };    // native cast -> hw cvt on gfx1250
#endif
  unsigned int u;
  __builtin_memcpy(&u, &r, 4);
  return u;
}
static __device__ __forceinline__ unsigned short f32_to_bf16_s(float x) {
  __bf16 h = (__bf16)x;
  unsigned short s;
  __builtin_memcpy(&s, &h, 2);
  return s;
}

#if __has_builtin(__builtin_amdgcn_tanhf)
#define HW_TANH(x) __builtin_amdgcn_tanhf(x)
#elif __has_builtin(__builtin_amdgcn_tanh_f32)
#define HW_TANH(x) __builtin_amdgcn_tanh_f32(x)
#else
#define HW_TANH(x) tanhf(x)
#endif

// ---------------- async global->LDS 16B copy (guarded) ---------------------
#if __has_builtin(__builtin_amdgcn_global_load_async_to_lds_b128)
#define HAVE_ASYNC_LDS 1
#else
#define HAVE_ASYNC_LDS 0
#endif

static __device__ __forceinline__ void copy16_g2l(const unsigned short* g,
                                                  unsigned short* l) {
#if HAVE_ASYNC_LDS
  __builtin_amdgcn_global_load_async_to_lds_b128(
      (gptr_v4i)g, (lptr_v4i)l, 0, 0);
#else
  *(u32x4*)l = *(const u32x4*)g;
#endif
}
static __device__ __forceinline__ void wait_async_all() {
#if HAVE_ASYNC_LDS
#if __has_builtin(__builtin_amdgcn_s_wait_asynccnt)
  __builtin_amdgcn_s_wait_asynccnt(0);
#else
  asm volatile("s_wait_asynccnt 0" ::: "memory");
#endif
#endif
}

// Load one 16x32 bf16 A/B fragment from LDS (row-major, K contiguous, row stride 32).
// ISA layout: lanes 0-15 -> rows 0-15 with K=[0..7] in v0-3, K=[16..23] in v4-7;
// lanes 16-31 -> same rows with K=[8..15], K=[24..31].  => two ds_load_b128 per lane.
static __device__ __forceinline__ v16bf ld_frag(const unsigned short* lds, int row, int kbase) {
  v16bf f;
  u32x4* p = (u32x4*)&f;
  const unsigned short* base = lds + row * 32 + kbase;
  p[0] = *(const u32x4*)(base);       // K = kbase .. kbase+7
  p[1] = *(const u32x4*)(base + 16);  // K = kbase+16 .. kbase+23
  return f;
}

// ---------------------------------------------------------------------------
// Prep: fp32 [R][C] row-major  ->  bf16 [C][R] (transposed), tiled via LDS.
// ---------------------------------------------------------------------------
__global__ __launch_bounds__(256) void ode_transpose_bf16(
    const float* __restrict__ in, unsigned short* __restrict__ out, int R, int C)
{
  __shared__ float tile[32][33];
  const int tx = threadIdx.x & 31;
  const int ty = threadIdx.x >> 5;          // 0..7
  const int c0 = blockIdx.x * 32;
  const int r0 = blockIdx.y * 32;
#pragma unroll
  for (int j = 0; j < 4; ++j)
    tile[ty + j * 8][tx] = in[(size_t)(r0 + ty + j * 8) * C + (c0 + tx)];
  __syncthreads();
#pragma unroll
  for (int j = 0; j < 4; ++j) {
    int cc = c0 + ty + j * 8;
    out[(size_t)cc * R + (r0 + tx)] = f32_to_bf16_s(tile[tx][ty + j * 8]);
  }
}

// ---------------------------------------------------------------------------
// Prep: flat fp32 -> bf16 (for z0 shadow copy).  n must be /4 per thread.
// ---------------------------------------------------------------------------
__global__ __launch_bounds__(256) void ode_cvt_bf16(
    const float* __restrict__ in, unsigned short* __restrict__ out, int n4)
{
  int i = blockIdx.x * 256 + threadIdx.x;
  if (i < n4) {
    const float4 v = *(const float4*)(in + (size_t)i * 4);
    unsigned int lo = pack2_bf16(v.x, v.y);
    unsigned int hi = pack2_bf16(v.z, v.w);
    uint2 p; p.x = lo; p.y = hi;
    *(uint2*)(out + (size_t)i * 4) = p;
  }
}

// ---------------------------------------------------------------------------
// GEMM1: H = tanh(Zbf @ W1 + b1 + ti*wt)   Zbf bf16 [BS][DD], W1T bf16 [HIDD][DD]
// Output H bf16 [BS][HIDD].
// Block tile 64(M) x 64(N), 128 threads = 4 waves, wave tile 32x32.
// Double-buffered LDS; all staging is pure async bf16 copy.
// ---------------------------------------------------------------------------
__global__ __launch_bounds__(128) void ode_gemm1_tanh(
    const unsigned short* __restrict__ Zbf,
    const unsigned short* __restrict__ W1T,
    const float* __restrict__ b1,
    const float* __restrict__ wt,
    const float* __restrict__ t, int step,
    unsigned short* __restrict__ H)
{
  const int seg = step >> 1, sub = step & 1;
  const float t0 = t[seg], t1 = t[seg + 1];
  const float hstep = (t1 - t0) * 0.5f;
  const float ti = t0 + hstep * (float)sub;

  __shared__ unsigned short Alds[2][64 * 32];   // M x K  (bf16, K contiguous)
  __shared__ unsigned short Blds[2][64 * 32];   // N x K  (bf16, K contiguous)

  const int tid      = threadIdx.x;
  const int lane     = tid & 31;
  const int wid      = tid >> 5;
  const int waveM    = (wid & 1) * 32;
  const int waveN    = (wid >> 1) * 32;
  const int laneHalf = lane >> 4;
  const int laneIdx  = lane & 15;
  const int kbase    = laneHalf * 8;

  const int n0 = blockIdx.x * 64;            // over HIDD
  const int m0 = blockIdx.y * 64;            // over BS

  v8f acc[2][2];
#pragma unroll
  for (int i = 0; i < 2; ++i)
#pragma unroll
    for (int j = 0; j < 2; ++j) acc[i][j] = {};

  auto stage = [&](int k0, int buf) {        // both tiles: pure bf16 16B copies
#pragma unroll
    for (int i = 0; i < 2; ++i) {
      int c = tid + i * 128;
      int r = c >> 2;
      int o = (c & 3) * 8;                   // element offset (8 ushorts = 16B)
      copy16_g2l(Zbf + (size_t)(m0 + r) * DD + k0 + o, &Alds[buf][r * 32 + o]);
      copy16_g2l(W1T + (size_t)(n0 + r) * DD + k0 + o, &Blds[buf][r * 32 + o]);
    }
  };

  const int nK = DD / 32;
  stage(0, 0);
  int cur = 0;
  for (int kc = 0; kc < nK; ++kc) {
    wait_async_all();
    __syncthreads();                         // staged data in `cur` visible
    if (kc + 1 < nK) {                       // overlap next-chunk staging
      __builtin_prefetch(W1T + (size_t)(n0 + (tid >> 1)) * DD + kc * 32 + 64, 0, 1);
      stage((kc + 1) * 32, cur ^ 1);
    }
    v16bf afrag[2], bfrag[2];
#pragma unroll
    for (int mt = 0; mt < 2; ++mt)
      afrag[mt] = ld_frag(Alds[cur], waveM + mt * 16 + laneIdx, kbase);
#pragma unroll
    for (int nt = 0; nt < 2; ++nt)
      bfrag[nt] = ld_frag(Blds[cur], waveN + nt * 16 + laneIdx, kbase);
#pragma unroll
    for (int mt = 0; mt < 2; ++mt)
#pragma unroll
      for (int nt = 0; nt < 2; ++nt)
        acc[mt][nt] = __builtin_amdgcn_wmma_f32_16x16x32_bf16(
            false, afrag[mt], false, bfrag[nt], (short)0, acc[mt][nt], false, false);
    cur ^= 1;
  }

  // Epilogue: tanh(acc + b1 + ti*wt) -> bf16 H
#pragma unroll
  for (int nt = 0; nt < 2; ++nt) {
    const int n = n0 + waveN + nt * 16 + laneIdx;
    const float bias = b1[n] + ti * wt[n];
#pragma unroll
    for (int mt = 0; mt < 2; ++mt) {
#pragma unroll
      for (int r = 0; r < 8; ++r) {
        const int m = m0 + waveM + mt * 16 + laneHalf * 8 + r;
        const float v = acc[mt][nt][r] + bias;
        H[(size_t)m * HIDD + n] = f32_to_bf16_s(HW_TANH(v));
      }
    }
  }
}

// ---------------------------------------------------------------------------
// GEMM2: Zout = Zin + h*(H @ W2 + b2);  also writes bf16 shadow Zbf of Zout.
// H bf16 [BS][HIDD], W2T bf16 [DD][HIDD]
// ---------------------------------------------------------------------------
__global__ __launch_bounds__(128) void ode_gemm2_update(
    const unsigned short* __restrict__ Hb,
    const unsigned short* __restrict__ W2T,
    const float* __restrict__ b2,
    const float* __restrict__ t, int step,
    const float* __restrict__ Zin,
    float* __restrict__ Zout,
    unsigned short* __restrict__ Zbf_out)
{
  const int seg = step >> 1;
  const float t0 = t[seg], t1 = t[seg + 1];
  const float hstep = (t1 - t0) * 0.5f;

  __shared__ unsigned short Alds[2][64 * 32];
  __shared__ unsigned short Blds[2][64 * 32];

  const int tid      = threadIdx.x;
  const int lane     = tid & 31;
  const int wid      = tid >> 5;
  const int waveM    = (wid & 1) * 32;
  const int waveN    = (wid >> 1) * 32;
  const int laneHalf = lane >> 4;
  const int laneIdx  = lane & 15;
  const int kbase    = laneHalf * 8;

  const int n0 = blockIdx.x * 64;            // over DD
  const int m0 = blockIdx.y * 64;            // over BS

  v8f acc[2][2];
#pragma unroll
  for (int i = 0; i < 2; ++i)
#pragma unroll
    for (int j = 0; j < 2; ++j) acc[i][j] = {};

  auto stage = [&](int k0, int buf) {
#pragma unroll
    for (int i = 0; i < 2; ++i) {
      int c = tid + i * 128;
      int r = c >> 2;
      int o = (c & 3) * 8;
      copy16_g2l(Hb  + (size_t)(m0 + r) * HIDD + k0 + o, &Alds[buf][r * 32 + o]);
      copy16_g2l(W2T + (size_t)(n0 + r) * HIDD + k0 + o, &Blds[buf][r * 32 + o]);
    }
  };

  const int nK = HIDD / 32;
  stage(0, 0);
  int cur = 0;
  for (int kc = 0; kc < nK; ++kc) {
    wait_async_all();
    __syncthreads();
    if (kc + 1 < nK) {
      __builtin_prefetch(W2T + (size_t)(n0 + (tid >> 1)) * HIDD + kc * 32 + 64, 0, 1);
      stage((kc + 1) * 32, cur ^ 1);
    }
    v16bf afrag[2], bfrag[2];
#pragma unroll
    for (int mt = 0; mt < 2; ++mt)
      afrag[mt] = ld_frag(Alds[cur], waveM + mt * 16 + laneIdx, kbase);
#pragma unroll
    for (int nt = 0; nt < 2; ++nt)
      bfrag[nt] = ld_frag(Blds[cur], waveN + nt * 16 + laneIdx, kbase);
#pragma unroll
    for (int mt = 0; mt < 2; ++mt)
#pragma unroll
      for (int nt = 0; nt < 2; ++nt)
        acc[mt][nt] = __builtin_amdgcn_wmma_f32_16x16x32_bf16(
            false, afrag[mt], false, bfrag[nt], (short)0, acc[mt][nt], false, false);
    cur ^= 1;
  }

  // Epilogue: Euler update in fp32 + bf16 shadow for next step's GEMM1
#pragma unroll
  for (int nt = 0; nt < 2; ++nt) {
    const int n = n0 + waveN + nt * 16 + laneIdx;
    const float bias = b2[n];
#pragma unroll
    for (int mt = 0; mt < 2; ++mt) {
#pragma unroll
      for (int r = 0; r < 8; ++r) {
        const int m = m0 + waveM + mt * 16 + laneHalf * 8 + r;
        const float v = acc[mt][nt][r] + bias;
        const float z = Zin[(size_t)m * DD + n] + hstep * v;
        Zout[(size_t)m * DD + n] = z;
        Zbf_out[(size_t)m * DD + n] = f32_to_bf16_s(z);
      }
    }
  }
}

// ---------------------------------------------------------------------------
extern "C" void kernel_launch(void* const* d_in, const int* in_sizes, int n_in,
                              void* d_out, int out_size, void* d_ws, size_t ws_size,
                              hipStream_t stream)
{
  const float* z0 = (const float*)d_in[0];
  const float* t  = (const float*)d_in[1];
  const float* W1 = (const float*)d_in[2];
  const float* b1 = (const float*)d_in[3];
  const float* wt = (const float*)d_in[4];
  const float* W2 = (const float*)d_in[5];
  const float* b2 = (const float*)d_in[6];
  float* out = (float*)d_out;

  // Workspace layout (bytes):
  //   W1T bf16 [HIDD][DD] : 4 MB
  //   W2T bf16 [DD][HIDD] : 4 MB
  //   H   bf16 [BS][HIDD] : 2 MB
  //   zb0 f32  [BS][DD]   : 2 MB
  //   zb1 f32  [BS][DD]   : 2 MB
  //   Zbf bf16 [BS][DD]   : 1 MB
  char* ws = (char*)d_ws;
  unsigned short* W1T  = (unsigned short*)ws;
  unsigned short* W2T  = (unsigned short*)(ws + (size_t)HIDD * DD * 2);
  unsigned short* Hbuf = (unsigned short*)(ws + (size_t)HIDD * DD * 4);
  float* zb0 = (float*)(ws + (size_t)HIDD * DD * 4 + (size_t)BS * HIDD * 2);
  float* zb1 = zb0 + (size_t)BS * DD;
  unsigned short* Zbf = (unsigned short*)(zb1 + (size_t)BS * DD);

  // One-time (per launch): bf16 transposed weights + bf16 shadow of z0
  ode_transpose_bf16<<<dim3(HIDD / 32, DD / 32), 256, 0, stream>>>(W1, W1T, DD, HIDD);
  ode_transpose_bf16<<<dim3(DD / 32, HIDD / 32), 256, 0, stream>>>(W2, W2T, HIDD, DD);
  {
    int n4 = (BS * DD) / 4;
    ode_cvt_bf16<<<(n4 + 255) / 256, 256, 0, stream>>>(z0, Zbf, n4);
  }

  const float* zc = z0;
  for (int s = 0; s < NSTEPS; ++s) {
    ode_gemm1_tanh<<<dim3(HIDD / 64, BS / 64), 128, 0, stream>>>(
        Zbf, W1T, b1, wt, t, s, Hbuf);
    float* zn = (s == NSTEPS - 1) ? out : ((s & 1) ? zb1 : zb0);
    ode_gemm2_update<<<dim3(DD / 64, BS / 64), 128, 0, stream>>>(
        Hbuf, W2T, b2, t, s, zc, zn, Zbf);
    zc = zn;
  }
}
